// RoPEMultiheadAttention_34462817583573
// MI455X (gfx1250) — matrix operand order repaired and checked by reading, not compiled, not run on hardware
//
#include <hip/hip_runtime.h>
#include <hip/hip_bf16.h>
#include <cstdint>
#include <cstddef>

// ---------------------------------------------------------------------------
// Types / WMMA helpers (CDNA5 / gfx1250, wave32)
// ---------------------------------------------------------------------------
typedef __bf16 bf16_t;
typedef __attribute__((ext_vector_type(16))) __bf16 v16bf;
typedef __attribute__((ext_vector_type(8)))  __bf16 v8bf;
typedef __attribute__((ext_vector_type(4)))  __bf16 v4bf;
typedef __attribute__((ext_vector_type(8)))  float  v8f;
typedef __attribute__((ext_vector_type(4)))  float  v4f;

__device__ __forceinline__ v8f wmma_bf16(v16bf a, v16bf b, v8f c) {
  // D = A(16x32 bf16) * B(32x16 bf16) + C(16x16 f32)
  return __builtin_amdgcn_wmma_f32_16x16x32_bf16(false, a, false, b,
                                                 (short)0, c, false, false);
}

// Problem dims
constexpr int BB = 2, SS = 2048, EE = 2048, HH = 16, DD = 128;

// ---------------------------------------------------------------------------
// CDNA5 async memory->LDS copy (ASYNCcnt path). 16B per lane.
//   LDS byte address in a VGPR; global address is an SGPR base plus a VGPR byte offset
// ---------------------------------------------------------------------------
__device__ __forceinline__ uint32_t lds_off(const void* p) {
  return (uint32_t)(uintptr_t)p;   // low 32 bits of LDS-aperture flat address
}
__device__ __forceinline__ void async_b128(uint32_t lds_byte, uint32_t gbyte,
                                           const void* base) {
  asm volatile("global_load_async_to_lds_b128 %0, %1, %2"
               :: "v"(lds_byte), "v"(gbyte), "s"(base) : "memory");
}
__device__ __forceinline__ void wait_async0() {
  asm volatile("s_wait_asynccnt 0x0" ::: "memory");
}

// ---------------------------------------------------------------------------
// Fragment loaders (layouts per CDNA5 ISA 7.12.2).
// A (16-bit 16x32): lanes 0-15: elems0..7=K0..7, 8..15=K16..23;
//                   lanes16-31: elems0..7=K8..15, 8..15=K24..31.
// B (16-bit 32x16) stored transposed in LDS [n][k]:
//   lanes 0-15: elems0..7=K0..7, 8..15=K16..23; lanes 16-31: K8..15, K24..31.
// ---------------------------------------------------------------------------
__device__ __forceinline__ v16bf load_A_frag(const bf16_t* lds, int m,
                                             int stride, int k0, int lane) {
  const bf16_t* p = lds + (m + (lane & 15)) * stride + k0 + ((lane >> 4) << 3);
  v8bf lo = *(const v8bf*)(p);
  v8bf hi = *(const v8bf*)(p + 16);
  return __builtin_shufflevector(lo, hi, 0,1,2,3,4,5,6,7,8,9,10,11,12,13,14,15);
}
__device__ __forceinline__ v16bf load_B_frag(const bf16_t* lds, int n,
                                             int stride, int k0, int lane) {
  const bf16_t* p = lds + (n + (lane & 15)) * stride + k0 + ((lane >> 4) << 3);
  v8bf lo = *(const v8bf*)(p);
  v8bf hi = *(const v8bf*)(p + 16);
  return __builtin_shufflevector(lo, hi, 0,1,2,3,4,5,6,7,8,9,10,11,12,13,14,15);
}

// ---------------------------------------------------------------------------
// GEMM tiling: 128x128 tile, K-step 32, 8 waves (4M x 2N), wave tile 32x64.
// ---------------------------------------------------------------------------
constexpr int TM = 128, TN = 128, TK = 32;
constexpr int LDA = TK + 8;   // 40 bf16 -> 80B rows (16B aligned)
constexpr int LDC = TN + 1;   // 129 floats

// x (f32) tile -> bf16 LDS [m][k];  W (f32) tile -> bf16 LDS transposed [n][k]
__device__ __forceinline__ void k1_load_tiles(const float* __restrict__ x,
                                              const float* __restrict__ W,
                                              bf16_t* As, bf16_t* Bs,
                                              int m0, int n0, int k0, int tid) {
#pragma unroll
  for (int i = 0; i < 4; ++i) {            // 1024 float4, 4/thread
    int idx = tid + i * 256;
    int r = idx >> 3;
    int c = (idx & 7) << 2;
    v4f xv = *(const v4f*)(x + (size_t)(m0 + r) * EE + k0 + c);
    v4bf pk;
#pragma unroll
    for (int j = 0; j < 4; ++j) pk[j] = (bf16_t)xv[j];
    *(v4bf*)(As + r * LDA + c) = pk;       // one ds_store_b64
  }
#pragma unroll
  for (int i = 0; i < 4; ++i) {            // unit = (n, 4 consecutive k)
    int idx = tid + i * 256;
    int n  = idx & 127;                    // lane-consecutive n -> coalesced
    int kc = idx >> 7;                     // 0..7
    const float* wp = W + (size_t)(k0 + kc * 4) * EE + n0 + n;
    v4bf pk;
#pragma unroll
    for (int j = 0; j < 4; ++j) pk[j] = (bf16_t)wp[(size_t)j * EE];
    *(v4bf*)(Bs + n * LDA + kc * 4) = pk;  // one ds_store_b64
  }
}

// ---------------------------------------------------------------------------
// Kernel 1: fused QKV projection + RoPE. grid (16 heads, 32 m-tiles, 3 {q,k,v})
// Double-buffered LDS, one barrier per K-step.
// ---------------------------------------------------------------------------
__global__ __launch_bounds__(256, 1)
void qkv_rope_gemm_kernel(const float* __restrict__ x,
                          const float* __restrict__ Wq, const float* __restrict__ bq,
                          const float* __restrict__ Wk, const float* __restrict__ bk,
                          const float* __restrict__ Wv, const float* __restrict__ bv,
                          bf16_t* __restrict__ qo, bf16_t* __restrict__ ko,
                          bf16_t* __restrict__ vo) {
  __shared__ __attribute__((aligned(16))) bf16_t As[2][TM * LDA];
  __shared__ __attribute__((aligned(16))) bf16_t Bs[2][TN * LDA];
  __shared__ __attribute__((aligned(16))) float  Cs[TM * LDC];

  const int tid  = threadIdx.x;
  const int lane = tid & 31;
  const int w    = tid >> 5;
  const int wm   = w >> 1;
  const int wn   = w & 1;
  const int h    = blockIdx.x;             // n-tile == head
  const int n0   = h * TN;
  const int m0   = blockIdx.y * TM;
  const int z    = blockIdx.z;

  const float* W    = (z == 0) ? Wq : (z == 1) ? Wk : Wv;
  const float* bias = (z == 0) ? bq : (z == 1) ? bk : bv;
  bf16_t*      outp = (z == 0) ? qo : (z == 1) ? ko : vo;

  v8f acc[2][4] = {};

  k1_load_tiles(x, W, As[0], Bs[0], m0, n0, 0, tid);
  __syncthreads();

  for (int kt = 0; kt < EE / TK; ++kt) {
    const int cur = kt & 1;
    if (kt + 1 < EE / TK)
      k1_load_tiles(x, W, As[cur ^ 1], Bs[cur ^ 1], m0, n0, (kt + 1) * TK, tid);

    v16bf a0 = load_A_frag(As[cur], wm * 32 + 0,  LDA, 0, lane);
    v16bf a1 = load_A_frag(As[cur], wm * 32 + 16, LDA, 0, lane);
#pragma unroll
    for (int nj = 0; nj < 4; ++nj) {
      v16bf bfr = load_B_frag(Bs[cur], wn * 64 + nj * 16, LDA, 0, lane);
      acc[0][nj] = wmma_bf16(a0, bfr, acc[0][nj]);
      acc[1][nj] = wmma_bf16(a1, bfr, acc[1][nj]);
    }
    __syncthreads();
  }

  // Stage accum (+bias) to LDS for cross-column RoPE pairing.
  const int colb = lane & 15;
  const int half = lane >> 4;
#pragma unroll
  for (int mi = 0; mi < 2; ++mi)
#pragma unroll
    for (int nj = 0; nj < 4; ++nj) {
      int col = wn * 64 + nj * 16 + colb;
      float bv_ = bias[n0 + col];
#pragma unroll
      for (int r = 0; r < 8; ++r) {
        int row = wm * 32 + mi * 16 + half * 8 + r;
        Cs[row * LDC + col] = acc[mi][nj][r] + bv_;
      }
    }
  __syncthreads();

  const bool do_rope = (z != 2);
  // -log2(10000)/64
  const float nl2 = -0.20762050593046938f;
  for (int e = tid; e < TM * TN; e += 256) {
    int row = e >> 7;
    int d   = e & 127;
    int mg  = m0 + row;
    int b   = mg >> 11;
    int s   = mg & 2047;
    float val = Cs[row * LDC + d];
    if (do_rope) {
      int i = d & 63;
      float inv = exp2f((float)i * nl2);   // 10000^(-i/64)
      float ang = (float)s * inv;
      float sn = __sinf(ang);
      float cn = __cosf(ang);
      float rot = (d < 64) ? -Cs[row * LDC + 2 * d + 1]
                           :  Cs[row * LDC + 2 * (d - 64)];
      val = val * cn + rot * sn;
    }
    outp[(((size_t)(b * HH + h)) * SS + s) * DD + d] = (bf16_t)val;
  }
}

// ---------------------------------------------------------------------------
// Kernel 2: flash-style attention per (b,h). grid (16 q-tiles, 32 b*h).
// Q/K tiles staged via async global->LDS (ASYNCcnt); V transposed with packed
// LDS stores; scores + PV on WMMA bf16; online softmax in f32.
// ---------------------------------------------------------------------------
constexpr int QT = 128, KT = 64;
constexpr int LQ = DD + 8;   // 136
constexpr int LK = DD + 8;
constexpr int LV = KT + 8;   // 72
constexpr int LP = KT + 8;
constexpr int LS = KT + 1;   // 65

__global__ __launch_bounds__(256, 1)
void attention_kernel(const bf16_t* __restrict__ qg, const bf16_t* __restrict__ kg,
                      const bf16_t* __restrict__ vg, const int* __restrict__ maskg,
                      bf16_t* __restrict__ ctx) {
  __shared__ __attribute__((aligned(16))) bf16_t Qs[QT * LQ];
  __shared__ __attribute__((aligned(16))) bf16_t Ks[KT * LK];
  __shared__ __attribute__((aligned(16))) bf16_t Vt[DD * LV];   // [d][key]
  __shared__ __attribute__((aligned(16))) float  Sb[QT * LS];
  __shared__ __attribute__((aligned(16))) bf16_t Ps[QT * LP];
  __shared__ float mrow[QT], lrow[QT], arow[QT];
  __shared__ float maskadd[KT];

  const int tid  = threadIdx.x;
  const int lane = tid & 31;
  const int w    = tid >> 5;
  const int wm   = w >> 1;
  const int wn   = w & 1;
  const int colb = lane & 15;
  const int half = lane >> 4;
  const int q0   = blockIdx.x * QT;
  const int bh   = blockIdx.y;
  const int b    = bh >> 4;
  const int h    = bh & 15;
  const size_t headoff = (size_t)bh * SS * DD;

  // Q tile via async copy: 2048 16B chunks, 8/thread.
#pragma unroll
  for (int i = 0; i < 8; ++i) {
    int idx = tid + i * 256;
    int r = idx >> 4;
    int c = (idx & 15) << 3;
    async_b128(lds_off(Qs + r * LQ + c),
               (uint32_t)((headoff + (size_t)(q0 + r) * DD + c) * sizeof(bf16_t)),
               qg);
  }
  if (tid < QT) { mrow[tid] = -3.0e38f; lrow[tid] = 0.0f; }

  v8f o[2][4] = {};
  const float isq = 0.08838834764831845f;  // 1/sqrt(128)

  for (int kb = 0; kb < SS / KT; ++kb) {
    const int kk0 = kb * KT;
    __syncthreads();
    // K block [key][d] via async copy: 1024 chunks, 4/thread.
#pragma unroll
    for (int i = 0; i < 4; ++i) {
      int idx = tid + i * 256;
      int r = idx >> 4;
      int c = (idx & 15) << 3;
      async_b128(lds_off(Ks + r * LK + c),
                 (uint32_t)((headoff + (size_t)(kk0 + r) * DD + c) * sizeof(bf16_t)),
                 kg);
    }
    // V transposed -> Vt[d][key]: unit (d, 4 keys), packed b64 stores.
#pragma unroll
    for (int i = 0; i < 8; ++i) {
      int idx = tid + i * 256;           // 2048 units
      int d  = idx & 127;                // lane-consecutive d -> coalesced
      int k4 = idx >> 7;                 // 0..15
      const bf16_t* vp = vg + headoff + (size_t)(kk0 + k4 * 4) * DD + d;
      v4bf pk;
#pragma unroll
      for (int j = 0; j < 4; ++j) pk[j] = vp[(size_t)j * DD];
      *(v4bf*)(Vt + d * LV + k4 * 4) = pk;
    }
    if (tid < KT)
      maskadd[tid] = (maskg[b * SS + kk0 + tid] != 0) ? 0.0f : -3.0e38f;
    wait_async0();
    __syncthreads();

    // S = Q K^T * isq : M=128 x N=64, K=128. Wave tile 32x32.
    v8f sc[2][2] = {};
#pragma unroll
    for (int ks = 0; ks < 4; ++ks) {
      v16bf a0 = load_A_frag(Qs, wm * 32 + 0,  LQ, ks * 32, lane);
      v16bf a1 = load_A_frag(Qs, wm * 32 + 16, LQ, ks * 32, lane);
      v16bf b0 = load_B_frag(Ks, wn * 32 + 0,  LK, ks * 32, lane);
      v16bf b1 = load_B_frag(Ks, wn * 32 + 16, LK, ks * 32, lane);
      sc[0][0] = wmma_bf16(a0, b0, sc[0][0]);
      sc[0][1] = wmma_bf16(a0, b1, sc[0][1]);
      sc[1][0] = wmma_bf16(a1, b0, sc[1][0]);
      sc[1][1] = wmma_bf16(a1, b1, sc[1][1]);
    }
#pragma unroll
    for (int mi = 0; mi < 2; ++mi)
#pragma unroll
      for (int nj = 0; nj < 2; ++nj) {
        int col = wn * 32 + nj * 16 + colb;
#pragma unroll
        for (int r = 0; r < 8; ++r) {
          int row = wm * 32 + mi * 16 + half * 8 + r;
          Sb[row * LS + col] = sc[mi][nj][r] * isq;
        }
      }
    __syncthreads();

    // Online softmax: one thread per query row.
    if (tid < QT) {
      int r = tid;
      float mx = mrow[r];
      float rmax = -3.0e38f;
      for (int j = 0; j < KT; ++j)
        rmax = fmaxf(rmax, Sb[r * LS + j] + maskadd[j]);
      float nm = fmaxf(mx, rmax);
      float al = __expf(mx - nm);
      float sum = 0.0f;
      for (int j = 0; j < KT; ++j) {
        float p = __expf(Sb[r * LS + j] + maskadd[j] - nm);
        Ps[r * LP + j] = (bf16_t)p;
        sum += p;
      }
      mrow[r] = nm;
      lrow[r] = lrow[r] * al + sum;
      arow[r] = al;
    }
    __syncthreads();

    // Rescale O accumulators by alpha(row), then O += P @ V.
#pragma unroll
    for (int mi = 0; mi < 2; ++mi)
#pragma unroll
      for (int r = 0; r < 8; ++r) {
        int row = wm * 32 + mi * 16 + half * 8 + r;
        float al = arow[row];
#pragma unroll
        for (int nj = 0; nj < 4; ++nj) o[mi][nj][r] *= al;
      }
#pragma unroll
    for (int ks = 0; ks < 2; ++ks) {
      v16bf a0 = load_A_frag(Ps, wm * 32 + 0,  LP, ks * 32, lane);
      v16bf a1 = load_A_frag(Ps, wm * 32 + 16, LP, ks * 32, lane);
#pragma unroll
      for (int nj = 0; nj < 4; ++nj) {
        v16bf bfr = load_B_frag(Vt, wn * 64 + nj * 16, LV, ks * 32, lane);
        o[0][nj] = wmma_bf16(a0, bfr, o[0][nj]);
        o[1][nj] = wmma_bf16(a1, bfr, o[1][nj]);
      }
    }
  }
  __syncthreads();

  // Normalize and store context bf16 at [b, s, h*D + d] (== [B,S,E]).
#pragma unroll
  for (int mi = 0; mi < 2; ++mi)
#pragma unroll
    for (int nj = 0; nj < 4; ++nj) {
      int d = wn * 64 + nj * 16 + colb;
#pragma unroll
      for (int r = 0; r < 8; ++r) {
        int row = wm * 32 + mi * 16 + half * 8 + r;
        float inv = 1.0f / lrow[row];
        ctx[((size_t)(b * SS + q0 + row)) * EE + h * DD + d] =
            (bf16_t)(o[mi][nj][r] * inv);
      }
    }
}

// ---------------------------------------------------------------------------
// Kernel 3: output projection ctx(bf16) @ Wo + bo -> f32.
// ctx tile async->LDS; Wo converted/packed; double-buffered.
// ---------------------------------------------------------------------------
__device__ __forceinline__ void k3_load_tiles(const bf16_t* __restrict__ ctx,
                                              const float* __restrict__ Wo,
                                              bf16_t* As, bf16_t* Bs,
                                              int m0, int n0, int k0, int tid) {
#pragma unroll
  for (int i = 0; i < 2; ++i) {            // 512 16B chunks, 2/thread
    int idx = tid + i * 256;
    int r = idx >> 2;
    int c = (idx & 3) << 3;
    async_b128(lds_off(As + r * LDA + c),
               (uint32_t)(((size_t)(m0 + r) * EE + k0 + c) * sizeof(bf16_t)),
               ctx);
  }
#pragma unroll
  for (int i = 0; i < 4; ++i) {
    int idx = tid + i * 256;
    int n  = idx & 127;
    int kc = idx >> 7;
    const float* wp = Wo + (size_t)(k0 + kc * 4) * EE + n0 + n;
    v4bf pk;
#pragma unroll
    for (int j = 0; j < 4; ++j) pk[j] = (bf16_t)wp[(size_t)j * EE];
    *(v4bf*)(Bs + n * LDA + kc * 4) = pk;
  }
}

__global__ __launch_bounds__(256, 1)
void out_proj_gemm_kernel(const bf16_t* __restrict__ ctx,
                          const float* __restrict__ Wo,
                          const float* __restrict__ bo,
                          float* __restrict__ out) {
  __shared__ __attribute__((aligned(16))) bf16_t As[2][TM * LDA];
  __shared__ __attribute__((aligned(16))) bf16_t Bs[2][TN * LDA];

  const int tid  = threadIdx.x;
  const int lane = tid & 31;
  const int w    = tid >> 5;
  const int wm   = w >> 1;
  const int wn   = w & 1;
  const int n0   = blockIdx.x * TN;
  const int m0   = blockIdx.y * TM;
  const int colb = lane & 15;
  const int half = lane >> 4;

  v8f acc[2][4] = {};

  k3_load_tiles(ctx, Wo, As[0], Bs[0], m0, n0, 0, tid);
  wait_async0();
  __syncthreads();

  for (int kt = 0; kt < EE / TK; ++kt) {
    const int cur = kt & 1;
    if (kt + 1 < EE / TK)
      k3_load_tiles(ctx, Wo, As[cur ^ 1], Bs[cur ^ 1], m0, n0, (kt + 1) * TK, tid);

    v16bf a0 = load_A_frag(As[cur], wm * 32 + 0,  LDA, 0, lane);
    v16bf a1 = load_A_frag(As[cur], wm * 32 + 16, LDA, 0, lane);
#pragma unroll
    for (int nj = 0; nj < 4; ++nj) {
      v16bf bfr = load_B_frag(Bs[cur], wn * 64 + nj * 16, LDA, 0, lane);
      acc[0][nj] = wmma_bf16(a0, bfr, acc[0][nj]);
      acc[1][nj] = wmma_bf16(a1, bfr, acc[1][nj]);
    }
    wait_async0();
    __syncthreads();
  }

#pragma unroll
  for (int mi = 0; mi < 2; ++mi)
#pragma unroll
    for (int nj = 0; nj < 4; ++nj) {
      int col = wn * 64 + nj * 16 + colb;
      float bb = bo[n0 + col];
#pragma unroll
      for (int r = 0; r < 8; ++r) {
        int row = wm * 32 + mi * 16 + half * 8 + r;
        out[(size_t)(m0 + row) * EE + n0 + col] = acc[mi][nj][r] + bb;
      }
    }
}

// ---------------------------------------------------------------------------
// Launch
// ---------------------------------------------------------------------------
extern "C" void kernel_launch(void* const* d_in, const int* in_sizes, int n_in,
                              void* d_out, int out_size, void* d_ws, size_t ws_size,
                              hipStream_t stream) {
  (void)in_sizes; (void)n_in; (void)out_size; (void)ws_size;
  const float* x    = (const float*)d_in[0];
  const int*   mask = (const int*)d_in[1];
  const float* Wq   = (const float*)d_in[2];
  const float* bq   = (const float*)d_in[3];
  const float* Wk   = (const float*)d_in[4];
  const float* bk   = (const float*)d_in[5];
  const float* Wv   = (const float*)d_in[6];
  const float* bv   = (const float*)d_in[7];
  const float* Wo   = (const float*)d_in[8];
  const float* bo   = (const float*)d_in[9];
  float* out = (float*)d_out;

  const size_t QE = (size_t)BB * HH * SS * DD;  // 8,388,608 elems
  bf16_t* qb  = (bf16_t*)d_ws;
  bf16_t* kb  = qb + QE;
  bf16_t* vb  = kb + QE;
  bf16_t* ctx = vb + QE;                        // 64 MB total workspace

  qkv_rope_gemm_kernel<<<dim3(EE / TN, (BB * SS) / TM, 3), 256, 0, stream>>>(
      x, Wq, bq, Wk, bk, Wv, bv, qb, kb, vb);
  attention_kernel<<<dim3(SS / QT, BB * HH), 256, 0, stream>>>(
      qb, kb, vb, mask, ctx);
  out_proj_gemm_kernel<<<dim3(EE / TN, (BB * SS) / TM), 256, 0, stream>>>(
      ctx, Wo, bo, out);
}
